// OptunaDynamicGraphSAGEModel_46265387712895
// MI455X (gfx1250) — matrix operand order repaired and checked by reading, not compiled, not run on hardware
//
#include <hip/hip_runtime.h>
#include <hip/hip_bf16.h>

// GraphSAGE (3 layers, mean aggregation) + linear head, for MI455X (gfx1250).
// Per layer:
//   1) zero agg
//   2) edge scatter: agg[dst] += h[src]       (wave-per-edge, f32 atomics, L2 resident)
//   3) fused WMMA GEMM: h' = relu(h @ Wself^T + b + (agg*invdeg) @ Wneigh^T)
//      K=256 via v_wmma_f32_16x16x32_f16 (f32 accumulate).
//      A panel (16x256, f16) staged once per block in LDS, shared by 8 waves.
//      B panel prepacked per layer into WMMA fragment layout (f16) -> pure b128 loads.
// Final: out = h @ fcW^T + fcb (wave-per-node dot product).

#define N_NODES   50000
#define N_EDGES   600000
#define FEATS     128
#define ROW_TILES (N_NODES / 16)   // 3125, exact
#define COL_TILES (FEATS / 16)     // 8
#define KCHUNKS   8                // 256 / 32

typedef __attribute__((ext_vector_type(16))) _Float16 v16h;
typedef __attribute__((ext_vector_type(8)))  float    v8f;

// ---------------------------------------------------------------- utilities
__global__ __launch_bounds__(256) void zero_f4(float4* __restrict__ p, int n4) {
    int i = blockIdx.x * blockDim.x + threadIdx.x;
    if (i < n4) p[i] = make_float4(0.f, 0.f, 0.f, 0.f);
}

__global__ __launch_bounds__(256) void count_deg(const int* __restrict__ dst,
                                                 float* __restrict__ deg, int nE) {
    int e = blockIdx.x * blockDim.x + threadIdx.x;
    if (e < nE) atomicAdd(&deg[dst[e]], 1.0f);
}

__global__ __launch_bounds__(256) void finalize_invdeg(float* __restrict__ deg, int n) {
    int i = blockIdx.x * blockDim.x + threadIdx.x;
    if (i < n) deg[i] = 1.0f / fmaxf(deg[i], 1.0f);
}

// ------------------------------------------------------- edge scatter (mean numerator)
__global__ __launch_bounds__(256) void scatter_add(const float* __restrict__ h,
                                                   const int* __restrict__ src,
                                                   const int* __restrict__ dst,
                                                   float* __restrict__ agg, int nE) {
    int t    = blockIdx.x * blockDim.x + threadIdx.x;
    int e    = t >> 5;
    int lane = t & 31;
    if (e >= nE) return;
    int s = src[e];
    int d = dst[e];
    float4 v = ((const float4*)(h + (size_t)s * FEATS))[lane];
    float* ag = agg + (size_t)d * FEATS + lane * 4;
    atomicAdd(ag + 0, v.x);
    atomicAdd(ag + 1, v.y);
    atomicAdd(ag + 2, v.z);
    atomicAdd(ag + 3, v.w);
}

// ------------------------------------------------------- weight prepack (once / layer)
// Pack B' = [Wself^T ; Wneigh^T] (K=256 x N=128) into WMMA B-fragment layout (f16):
//   packed[((kb*8 + c)*32 + L)*16 + j] = W'[o = c*16 + L%16][k = kb*32 + (L/16)*16 + j]
// 2048 fragment slots of 16 halfs; one thread per slot.
__global__ __launch_bounds__(256) void pack_weights(const float* __restrict__ Wself,
                                                    const float* __restrict__ Wneigh,
                                                    _Float16* __restrict__ packed) {
    int slot = blockIdx.x * blockDim.x + threadIdx.x;   // 0..2047
    if (slot >= KCHUNKS * COL_TILES * 32) return;
    int kb = slot >> 8;
    int c  = (slot >> 5) & 7;
    int L  = slot & 31;
    int m  = L & 15;
    int hf = L >> 4;
    const float* W  = (kb < 4) ? Wself : Wneigh;
    const float* wp = W + (size_t)(c * 16 + m) * FEATS + (kb & 3) * 32 + hf * 16;
    _Float16* dp = packed + (size_t)slot * 16;
#pragma unroll
    for (int j = 0; j < 16; ++j) dp[j] = (_Float16)wp[j];
}

// ------------------------------------------------------- fused dual-GEMM + bias + relu
// One block per 16-row output tile; wave w computes col-tile w.
// LDS holds the converted A panel as WMMA fragments: ldsA[kb][lane] (8 KB).
__global__ __launch_bounds__(256) void sage_gemm(const float* __restrict__ hin,
                                                 const float* __restrict__ agg,
                                                 const float* __restrict__ invdeg,
                                                 const _Float16* __restrict__ packedB,
                                                 const float* __restrict__ bias,
                                                 float* __restrict__ hout) {
    __shared__ v16h ldsA[KCHUNKS][32];

    int rowTile = blockIdx.x;
    int t    = threadIdx.x;
    int lane = t & 31;

    // ---- cooperative fill: thread t converts fragment slot (kb = t/32, lane = t%32)
    {
        int kb = t >> 5;              // wave-uniform
        int m  = lane & 15;
        int hf = lane >> 4;
        int arow = rowTile * 16 + m;
        const float* Asrc   = (kb < 4) ? hin : agg;
        float        ascale = (kb < 4) ? 1.0f : invdeg[arow];
        const float* ap = Asrc + (size_t)arow * FEATS + (kb & 3) * 32 + hf * 16;
        float4 a0 = ((const float4*)ap)[0];
        float4 a1 = ((const float4*)ap)[1];
        float4 a2 = ((const float4*)(ap + 8))[0];
        float4 a3 = ((const float4*)(ap + 8))[1];
        v16h av;
        av[0]  = (_Float16)(a0.x * ascale); av[1]  = (_Float16)(a0.y * ascale);
        av[2]  = (_Float16)(a0.z * ascale); av[3]  = (_Float16)(a0.w * ascale);
        av[4]  = (_Float16)(a1.x * ascale); av[5]  = (_Float16)(a1.y * ascale);
        av[6]  = (_Float16)(a1.z * ascale); av[7]  = (_Float16)(a1.w * ascale);
        av[8]  = (_Float16)(a2.x * ascale); av[9]  = (_Float16)(a2.y * ascale);
        av[10] = (_Float16)(a2.z * ascale); av[11] = (_Float16)(a2.w * ascale);
        av[12] = (_Float16)(a3.x * ascale); av[13] = (_Float16)(a3.y * ascale);
        av[14] = (_Float16)(a3.z * ascale); av[15] = (_Float16)(a3.w * ascale);
        ldsA[kb][lane] = av;          // 2x ds_store_b128
    }
    __syncthreads();

    // ---- WMMA main loop: 2 ds_load_b128 (A) + 2 global_load_b128 (B) + 1 wmma per chunk
    int colTile = t >> 5;
    const v16h* bfrag = (const v16h*)packedB + (size_t)colTile * 32 + lane;
    v8f acc = {};
#pragma unroll
    for (int kb = 0; kb < KCHUNKS; ++kb) {
        v16h a = ldsA[kb][lane];
        v16h b = bfrag[(size_t)kb * COL_TILES * 32];
        acc = __builtin_amdgcn_wmma_f32_16x16x32_f16(
            /*neg_a=*/false, a, /*neg_b=*/false, b,
            /*c_mod=*/(short)0, acc, /*reuse_a=*/false, /*reuse_b=*/false);
    }

    // ---- epilogue: D elem v -> row (v + 8*(lane/16)), col (lane%16); bias + relu
    int m   = lane & 15;
    int hf  = lane >> 4;
    float bv = bias[colTile * 16 + m];
#pragma unroll
    for (int v = 0; v < 8; ++v) {
        int orow  = rowTile * 16 + v + 8 * hf;
        float val = fmaxf(acc[v] + bv, 0.0f);
        hout[(size_t)orow * FEATS + colTile * 16 + m] = val;
    }
}

// ------------------------------------------------------- final 128 -> 1 projection
__global__ __launch_bounds__(256) void fc_head(const float* __restrict__ h,
                                               const float* __restrict__ fcW,
                                               const float* __restrict__ fcb,
                                               float* __restrict__ out, int n) {
    int t    = blockIdx.x * blockDim.x + threadIdx.x;
    int node = t >> 5;
    int lane = t & 31;
    if (node >= n) return;
    float4 a = ((const float4*)(h + (size_t)node * FEATS))[lane];
    float4 w = ((const float4*)fcW)[lane];
    float s = a.x * w.x + a.y * w.y + a.z * w.z + a.w * w.w;
#pragma unroll
    for (int off = 16; off > 0; off >>= 1) s += __shfl_xor(s, off, 32);
    if (lane == 0) out[node] = s + fcb[0];
}

// ---------------------------------------------------------------- launch
extern "C" void kernel_launch(void* const* d_in, const int* in_sizes, int n_in,
                              void* d_out, int out_size, void* d_ws, size_t ws_size,
                              hipStream_t stream) {
    const float* features = (const float*)d_in[0];
    const int*   src      = (const int*)d_in[1];
    const int*   dst      = (const int*)d_in[2];
    const float* Ws[3]    = {(const float*)d_in[3], (const float*)d_in[6], (const float*)d_in[9]};
    const float* Wn[3]    = {(const float*)d_in[4], (const float*)d_in[7], (const float*)d_in[10]};
    const float* bs[3]    = {(const float*)d_in[5], (const float*)d_in[8], (const float*)d_in[11]};
    const float* fcW      = (const float*)d_in[12];
    const float* fcb      = (const float*)d_in[13];
    float*       out      = (float*)d_out;

    // Workspace carve-up (all re-initialized every call; harness poisons ws).
    const size_t featBytes = (size_t)N_NODES * FEATS * sizeof(float);   // 25.6 MB
    const size_t packBytes = (size_t)KCHUNKS * COL_TILES * 32 * 16 * 2; // 64 KB / layer
    char*     ws     = (char*)d_ws;
    float*    invdeg = (float*)ws;
    float*    agg    = (float*)(ws + 204800);
    float*    hA     = (float*)(ws + 204800 + featBytes);
    float*    hB     = (float*)(ws + 204800 + 2 * featBytes);
    _Float16* pB[3];
    for (int l = 0; l < 3; ++l)
        pB[l] = (_Float16*)(ws + 204800 + 3 * featBytes + (size_t)l * packBytes);

    const int nF4  = (N_NODES * FEATS) / 4;
    const int zBlk = (nF4 + 255) / 256;
    const int dBlk = (N_EDGES + 255) / 256;
    const int nBlk = (N_NODES + 255) / 256;
    const int sBlk = (N_EDGES * 32 + 255) / 256;   // wave-per-edge scatter
    const int pBlk = (KCHUNKS * COL_TILES * 32 + 255) / 256;  // 8 blocks
    const int fBlk = (N_NODES * 32 + 255) / 256;

    // Degrees -> 1/max(deg,1)
    zero_f4<<<(N_NODES / 4 + 255) / 256, 256, 0, stream>>>((float4*)invdeg, N_NODES / 4);
    count_deg<<<dBlk, 256, 0, stream>>>(dst, invdeg, N_EDGES);
    finalize_invdeg<<<nBlk, 256, 0, stream>>>(invdeg, N_NODES);

    // Prepack all layer weights into WMMA B-fragment layout (f16)
    for (int l = 0; l < 3; ++l)
        pack_weights<<<pBlk, 256, 0, stream>>>(Ws[l], Wn[l], pB[l]);

    const float* hin = features;
    float* bufs[2] = {hA, hB};
    for (int l = 0; l < 3; ++l) {
        zero_f4<<<zBlk, 256, 0, stream>>>((float4*)agg, nF4);
        scatter_add<<<sBlk, 256, 0, stream>>>(hin, src, dst, agg, N_EDGES);
        float* hout = bufs[l & 1];
        sage_gemm<<<ROW_TILES, 256, 0, stream>>>(hin, agg, invdeg, pB[l], bs[l], hout);
        hin = hout;
    }
    fc_head<<<fBlk, 256, 0, stream>>>(hin, fcW, fcb, out, N_NODES);
}
